// GAT_5806795784462
// MI455X (gfx1250) — matrix operand order repaired and checked by reading, not compiled, not run on hardware
//
#include <hip/hip_runtime.h>
#include <hip/hip_bf16.h>

#define NN 100000
#define NE 800000

typedef __attribute__((ext_vector_type(16))) __bf16    v16bf;
typedef __attribute__((ext_vector_type(8)))  float     v8f;
typedef __attribute__((ext_vector_type(4)))  unsigned  u32x4;

union ABu { u32x4 u[2]; v16bf v; };

__device__ __forceinline__ unsigned short f2bf(float f) {
    unsigned u = __float_as_uint(f);
    u += 0x7FFFu + ((u >> 16) & 1u);   // round-to-nearest-even
    return (unsigned short)(u >> 16);
}

__device__ __forceinline__ float lrelu(float x) { return x > 0.f ? x : 0.2f * x; }

__device__ __forceinline__ void atomicMaxF(float* addr, float val) {
    // monotone bit trick: works with -inf init, mixed signs
    if (val >= 0.f) atomicMax((int*)addr, __float_as_int(val));
    else            atomicMin((unsigned int*)addr, (unsigned int)__float_as_int(val));
}

// ---------------- utility kernels ----------------
__global__ void k_fill(float* p, int n, float v) {
    int i = blockIdx.x * blockDim.x + threadIdx.x;
    if (i < n) p[i] = v;
}

__global__ void k_cvt_bf16(const float* __restrict__ x, unsigned short* __restrict__ xb, int n) {
    int i = blockIdx.x * blockDim.x + threadIdx.x;
    if (i < n) xb[i] = f2bf(x[i]);
}

// Pack row-major W[K][n] (ncols columns) into WMMA-B lane layout:
// Wp[((kb*NT+nt)*32 + lane)*16 + j] = W[(kb*32 + (j>=8?16:0) + (lane>=16?8:0) + (j&7))*ncols + nt*16 + (lane&15)]
__global__ void k_pack(const float* __restrict__ W, unsigned short* __restrict__ Wp,
                       int KB, int NT, int ncols) {
    int idx = blockIdx.x * blockDim.x + threadIdx.x;
    if (idx >= KB * NT * 512) return;
    int j    = idx & 15;
    int lane = (idx >> 4) & 31;
    int t    = idx >> 9;        // kb*NT + nt
    int nt   = t % NT;
    int kb   = t / NT;
    int n    = nt * 16 + (lane & 15);
    int K    = kb * 32 + ((j >> 3) << 4) + ((lane >> 4) << 3) + (j & 7);
    Wp[idx] = f2bf(W[K * ncols + n]);
}

// ---------------- GEMM 1: [N,128]x[128,128] (x2 matrices), bf16 WMMA ----------------
__global__ void __launch_bounds__(256)
k_gemm1(const unsigned short* __restrict__ xb,
        const unsigned short* __restrict__ wsp, const unsigned short* __restrict__ wdp,
        const float* __restrict__ bsv, const float* __restrict__ bdv,
        float* __restrict__ hs1, float* __restrict__ hd1) {
    int nt   = threadIdx.x >> 5;     // 8 waves -> 8 N-tiles
    int lane = threadIdx.x & 31;
    int r0   = blockIdx.x * 16;
    int hi   = lane >> 4;
    int m    = lane & 15;
    v8f cs = {}, cd = {};
#pragma unroll
    for (int kb = 0; kb < 4; ++kb) {
        ABu a, bs, bd;
        const u32x4* pa = (const u32x4*)(xb + (r0 + m) * 128 + kb * 32 + hi * 8);
        a.u[0] = pa[0]; a.u[1] = pa[2];                       // K groups: hi*8.., 16+hi*8..
        const u32x4* pbs = (const u32x4*)(wsp + (size_t)((kb * 8 + nt) * 32 + lane) * 16);
        bs.u[0] = pbs[0]; bs.u[1] = pbs[1];
        const u32x4* pbd = (const u32x4*)(wdp + (size_t)((kb * 8 + nt) * 32 + lane) * 16);
        bd.u[0] = pbd[0]; bd.u[1] = pbd[1];
        cs = __builtin_amdgcn_wmma_f32_16x16x32_bf16(false, a.v, false, bs.v, (short)0, cs, false, false);
        cd = __builtin_amdgcn_wmma_f32_16x16x32_bf16(false, a.v, false, bd.v, (short)0, cd, false, false);
    }
    int col = nt * 16 + m;
    float biass = bsv[col], biasd = bdv[col];
#pragma unroll
    for (int i = 0; i < 8; ++i) {
        int row = r0 + hi * 8 + i;
        hs1[row * 128 + col] = cs[i] + biass;
        hd1[row * 128 + col] = cd[i] + biasd;
    }
}

// ---------------- GEMM 2: [N,128]x[128,16] (x2), bf16 WMMA ----------------
__global__ void __launch_bounds__(256)
k_gemm2(const unsigned short* __restrict__ hb,
        const unsigned short* __restrict__ wsp, const unsigned short* __restrict__ wdp,
        const float* __restrict__ bsv, const float* __restrict__ bdv,
        float* __restrict__ hs2, float* __restrict__ hd2) {
    int w = (blockIdx.x * blockDim.x + threadIdx.x) >> 5;
    if (w >= NN / 16) return;
    int lane = threadIdx.x & 31;
    int hi = lane >> 4, m = lane & 15;
    int r0 = w * 16;
    v8f cs = {}, cd = {};
#pragma unroll
    for (int kb = 0; kb < 4; ++kb) {
        ABu a, bs, bd;
        const u32x4* pa = (const u32x4*)(hb + (r0 + m) * 128 + kb * 32 + hi * 8);
        a.u[0] = pa[0]; a.u[1] = pa[2];
        const u32x4* pbs = (const u32x4*)(wsp + (size_t)(kb * 32 + lane) * 16);
        bs.u[0] = pbs[0]; bs.u[1] = pbs[1];
        const u32x4* pbd = (const u32x4*)(wdp + (size_t)(kb * 32 + lane) * 16);
        bd.u[0] = pbd[0]; bd.u[1] = pbd[1];
        cs = __builtin_amdgcn_wmma_f32_16x16x32_bf16(false, a.v, false, bs.v, (short)0, cs, false, false);
        cd = __builtin_amdgcn_wmma_f32_16x16x32_bf16(false, a.v, false, bd.v, (short)0, cd, false, false);
    }
    float biass = bsv[m], biasd = bdv[m];
#pragma unroll
    for (int i = 0; i < 8; ++i) {
        int row = r0 + hi * 8 + i;
        hs2[row * 16 + m] = cs[i] + biass;
        hd2[row * 16 + m] = cd[i] + biasd;
    }
}

// ---------------- layer-1 edge passes (H=4, D=32) ----------------
__global__ void __launch_bounds__(256)
k_score1(const float* __restrict__ hs1, const float* __restrict__ hd1,
         const int* __restrict__ ei, const float* __restrict__ attn1,
         float* __restrict__ score1, float* __restrict__ m1) {
    int e    = (blockIdx.x * blockDim.x + threadIdx.x) >> 5;  // wave per edge
    int lane = threadIdx.x & 31;
    if (e >= NE) return;
    int src = ei[e], dst = ei[NE + e];
    float4 a = ((const float4*)(hs1 + src * 128))[lane];
    float4 b = ((const float4*)(hd1 + dst * 128))[lane];
    float4 w = ((const float4*)attn1)[lane];
    float p = lrelu(a.x + b.x) * w.x + lrelu(a.y + b.y) * w.y +
              lrelu(a.z + b.z) * w.z + lrelu(a.w + b.w) * w.w;
    p += __shfl_xor(p, 1, 32);
    p += __shfl_xor(p, 2, 32);
    p += __shfl_xor(p, 4, 32);      // sum over 8-lane head group (32 dims)
    if ((lane & 7) == 0) {
        int h = lane >> 3;
        score1[e * 4 + h] = p;
        atomicMaxF(m1 + dst * 4 + h, p);
    }
}

__global__ void __launch_bounds__(256)
k_expden1(const int* __restrict__ ei, float* __restrict__ score1,
          const float* __restrict__ m1, float* __restrict__ den1) {
    int idx = blockIdx.x * blockDim.x + threadIdx.x;
    if (idx >= 4 * NE) return;
    int e = idx >> 2, h = idx & 3;
    int dst = ei[NE + e];
    float ex = __expf(score1[idx] - m1[dst * 4 + h]);
    score1[idx] = ex;
    atomicAdd(den1 + dst * 4 + h, ex);
}

__global__ void __launch_bounds__(256)
k_agg1(const float* __restrict__ hs1, const int* __restrict__ ei,
       const float* __restrict__ ex, const float* __restrict__ den1,
       float* __restrict__ agg1) {
    int e    = (blockIdx.x * blockDim.x + threadIdx.x) >> 5;
    int lane = threadIdx.x & 31;
    if (e >= NE) return;
    int src = ei[e], dst = ei[NE + e];
    int h = lane >> 3;
    float alpha = ex[e * 4 + h] / den1[dst * 4 + h];
    float4 hv = ((const float4*)(hs1 + src * 128))[lane];
    float* o = agg1 + dst * 128 + lane * 4;
    atomicAdd(o + 0, alpha * hv.x);
    atomicAdd(o + 1, alpha * hv.y);
    atomicAdd(o + 2, alpha * hv.z);
    atomicAdd(o + 3, alpha * hv.w);
}

__global__ void k_elu_bf16(const float* __restrict__ agg, unsigned short* __restrict__ hb, int n) {
    int i = blockIdx.x * blockDim.x + threadIdx.x;
    if (i >= n) return;
    float v = agg[i];
    hb[i] = f2bf(v > 0.f ? v : __expf(v) - 1.f);
}

// ---------------- layer-2 edge passes (H=1, D=16) ----------------
__global__ void __launch_bounds__(256)
k_score2(const float* __restrict__ hs2, const float* __restrict__ hd2,
         const int* __restrict__ ei, const float* __restrict__ attn2,
         float* __restrict__ score2, float* __restrict__ m2) {
    int e = blockIdx.x * blockDim.x + threadIdx.x;
    if (e >= NE) return;
    int src = ei[e], dst = ei[NE + e];
    const float4* ps = (const float4*)(hs2 + src * 16);
    const float4* pd = (const float4*)(hd2 + dst * 16);
    const float4* pw = (const float4*)attn2;
    float s = 0.f;
#pragma unroll
    for (int q = 0; q < 4; ++q) {
        float4 a = ps[q], b = pd[q], w = pw[q];
        s += lrelu(a.x + b.x) * w.x + lrelu(a.y + b.y) * w.y +
             lrelu(a.z + b.z) * w.z + lrelu(a.w + b.w) * w.w;
    }
    score2[e] = s;
    atomicMaxF(m2 + dst, s);
}

__global__ void __launch_bounds__(256)
k_expden2(const int* __restrict__ ei, float* __restrict__ score2,
          const float* __restrict__ m2, float* __restrict__ den2) {
    int e = blockIdx.x * blockDim.x + threadIdx.x;
    if (e >= NE) return;
    int dst = ei[NE + e];
    float ex = __expf(score2[e] - m2[dst]);
    score2[e] = ex;
    atomicAdd(den2 + dst, ex);
}

__global__ void __launch_bounds__(256)
k_agg2(const float* __restrict__ hs2, const int* __restrict__ ei,
       const float* __restrict__ ex, const float* __restrict__ den2,
       float* __restrict__ out) {
    int idx = blockIdx.x * blockDim.x + threadIdx.x;
    if (idx >= 4 * NE) return;
    int e = idx >> 2, q = idx & 3;
    int src = ei[e], dst = ei[NE + e];
    float alpha = ex[e] / den2[dst];
    float4 hv = ((const float4*)(hs2 + src * 16))[q];
    float* o = out + dst * 16 + q * 4;
    atomicAdd(o + 0, alpha * hv.x);
    atomicAdd(o + 1, alpha * hv.y);
    atomicAdd(o + 2, alpha * hv.z);
    atomicAdd(o + 3, alpha * hv.w);
}

// ---------------- launcher ----------------
extern "C" void kernel_launch(void* const* d_in, const int* in_sizes, int n_in,
                              void* d_out, int out_size, void* d_ws, size_t ws_size,
                              hipStream_t stream) {
    const float* feat  = (const float*)d_in[0];
    const int*   ei    = (const int*)  d_in[1];
    const float* W1s   = (const float*)d_in[2];
    const float* b1s   = (const float*)d_in[3];
    const float* W1d   = (const float*)d_in[4];
    const float* b1d   = (const float*)d_in[5];
    const float* attn1 = (const float*)d_in[6];
    const float* W2s   = (const float*)d_in[7];
    const float* b2s   = (const float*)d_in[8];
    const float* W2d   = (const float*)d_in[9];
    const float* b2d   = (const float*)d_in[10];
    const float* attn2 = (const float*)d_in[11];
    float* out = (float*)d_out;

    char* ws = (char*)d_ws;
    size_t off = 0;
    auto carve = [&](size_t bytes) -> char* {
        char* p = ws + off;
        off += (bytes + 255) & ~(size_t)255;
        return p;
    };
    unsigned short* xb   = (unsigned short*)carve((size_t)NN * 128 * 2); // reused as hb after GEMM1
    unsigned short* w1sp = (unsigned short*)carve(16384 * 2);
    unsigned short* w1dp = (unsigned short*)carve(16384 * 2);
    unsigned short* w2sp = (unsigned short*)carve(2048 * 2);
    unsigned short* w2dp = (unsigned short*)carve(2048 * 2);
    float* hs1 = (float*)carve((size_t)NN * 128 * 4);
    char*  l2r = carve((size_t)NN * 128 * 4);   // hd1 during layer 1; layer-2 buffers after
    float* hd1 = (float*)l2r;
    float* score1 = (float*)carve((size_t)NE * 4 * 4);
    float* m1   = (float*)carve((size_t)NN * 4 * 4);
    float* den1 = (float*)carve((size_t)NN * 4 * 4);
    float* agg1 = (float*)carve((size_t)NN * 128 * 4);
    // overlay inside l2r (hd1 is dead after k_score1)
    float* hs2    = (float*)(l2r);
    float* hd2    = (float*)(l2r + (8u << 20));
    float* score2 = (float*)(l2r + (16u << 20));
    float* m2     = (float*)(l2r + (20u << 20));
    float* den2   = (float*)(l2r + (21u << 20));
    unsigned short* hb = xb;

    const float ninf = -__builtin_huge_valf();

    // init (layer-1 + output)
    k_fill<<<(NN * 4 + 255) / 256, 256, 0, stream>>>(m1, NN * 4, ninf);
    k_fill<<<(NN * 4 + 255) / 256, 256, 0, stream>>>(den1, NN * 4, 0.f);
    k_fill<<<(NN * 128 + 255) / 256, 256, 0, stream>>>(agg1, NN * 128, 0.f);
    k_fill<<<(NN * 16 + 255) / 256, 256, 0, stream>>>(out, NN * 16, 0.f);

    // convert + pack
    k_cvt_bf16<<<(NN * 128 + 255) / 256, 256, 0, stream>>>(feat, xb, NN * 128);
    k_pack<<<(16384 + 255) / 256, 256, 0, stream>>>(W1s, w1sp, 4, 8, 128);
    k_pack<<<(16384 + 255) / 256, 256, 0, stream>>>(W1d, w1dp, 4, 8, 128);
    k_pack<<<(2048 + 255) / 256, 256, 0, stream>>>(W2s, w2sp, 4, 1, 16);
    k_pack<<<(2048 + 255) / 256, 256, 0, stream>>>(W2d, w2dp, 4, 1, 16);

    // layer 1
    k_gemm1<<<NN / 16, 256, 0, stream>>>(xb, w1sp, w1dp, b1s, b1d, hs1, hd1);
    k_score1<<<NE / 8, 256, 0, stream>>>(hs1, hd1, ei, attn1, score1, m1);
    k_expden1<<<(4 * NE + 255) / 256, 256, 0, stream>>>(ei, score1, m1, den1);
    k_agg1<<<NE / 8, 256, 0, stream>>>(hs1, ei, score1, den1, agg1);
    k_elu_bf16<<<(NN * 128 + 255) / 256, 256, 0, stream>>>(agg1, hb, NN * 128);

    // layer-2 init (after hd1 dead)
    k_fill<<<(NN + 255) / 256, 256, 0, stream>>>(m2, NN, ninf);
    k_fill<<<(NN + 255) / 256, 256, 0, stream>>>(den2, NN, 0.f);

    // layer 2
    k_gemm2<<<((NN / 16) * 32 + 255) / 256, 256, 0, stream>>>(hb, w2sp, w2dp, b2s, b2d, hs2, hd2);
    k_score2<<<(NE + 255) / 256, 256, 0, stream>>>(hs2, hd2, ei, attn2, score2, m2);
    k_expden2<<<(NE + 255) / 256, 256, 0, stream>>>(ei, score2, m2, den2);
    k_agg2<<<(4 * NE + 255) / 256, 256, 0, stream>>>(hs2, ei, score2, den2, out);
}
